// TiedEmbeddingLinear_42992622633478
// MI455X (gfx1250) — compile-verified, matrix operand order checked by reading, not run on hardware
//
#include <hip/hip_runtime.h>
#include <cmath>

// CDNA5 (gfx1250) wave32 WMMA types
typedef __attribute__((ext_vector_type(16))) _Float16 half16v;  // WMMA A/B fragment
typedef __attribute__((ext_vector_type(2)))  __fp16   fp16x2;   // cvt_pkrtz result type
typedef __attribute__((ext_vector_type(8)))  float    f32x8;    // WMMA C/D fragment

#define BM 128          // block tile M
#define BN 128          // block tile N (vocab)
#define BK 64           // block tile K per LDS stage (2 WMMA K-steps)
#define LDK 72          // padded LDS row stride in f16 elements (BK + 8)

__global__ __launch_bounds__(256)
void qgemm_wmma_f16(const float* __restrict__ x,     // [M, K] fp32
                    const int*   __restrict__ qw,    // [N, K] int32 codes 0..255
                    const float* __restrict__ absmax,// [N, K/256] fp32
                    float*       __restrict__ out,   // [M, N] fp32
                    int M, int N, int K)
{
  __shared__ __align__(16) _Float16 As[2][BM][LDK];
  __shared__ __align__(16) _Float16 Bs[2][BN][LDK];

  const int t       = threadIdx.x;
  const int wave    = t >> 5;          // 8 wave32s
  const int lane    = t & 31;
  const int laneLo  = lane & 15;
  const int laneHi8 = (lane >> 4) << 3;   // K sub-offset 0 or 8 (ISA 16-bit A/B layout)

  const int m0 = blockIdx.x * BM;      // M-tile fast-varying -> weight stripe L2 reuse
  const int n0 = blockIdx.y * BN;

  // wave grid 2(M) x 4(N): each wave computes 64M x 32N = 4x2 WMMA tiles
  const int wmBase = (wave >> 2) * 64;
  const int wnBase = (wave & 3)  * 32;

  // global->LDS staging: 2 threads per row, 32 contiguous K each
  const int rowL = t >> 1;             // 0..127
  const int colL = (t & 1) * 32;       // 0 or 32
  const float* xPtr  = x  + (size_t)(m0 + rowL) * K + colL;
  const int*   qPtr  = qw + (size_t)(n0 + rowL) * K + colL;
  const float* amRow = absmax + (size_t)(n0 + rowL) * (K >> 8);

  f32x8 acc[4][2];
  const f32x8 zero = {0.f, 0.f, 0.f, 0.f, 0.f, 0.f, 0.f, 0.f};
  #pragma unroll
  for (int i = 0; i < 4; ++i)
    #pragma unroll
    for (int j = 0; j < 2; ++j)
      acc[i][j] = zero;

  float4 aReg[8];   // staged x (32 fp32)
  int4   bReg[8];   // staged codes (32 codes)
  float  scNext;    // staged block scale (prefetched with g2r)

  auto g2r = [&](int s) {
    const int k0 = s * BK;
    #pragma unroll
    for (int i = 0; i < 8; ++i) {
      aReg[i] = *(const float4*)(xPtr + k0 + i * 4);
      bReg[i] = *(const int4*)  (qPtr + k0 + i * 4);
    }
    scNext = amRow[k0 >> 8];           // one 64-chunk never crosses a 256-block
  };

  auto r2lds = [&](int buf) {
    const float sc  = scNext * (1.0f / 127.5f);
    const float ofs = -127.5f * sc;                    // w = code*sc + ofs
    const fp16x2 sc2  = {(__fp16)sc,  (__fp16)sc};
    const fp16x2 ofs2 = {(__fp16)ofs, (__fp16)ofs};
    _Float16* aDst = &As[buf][rowL][colL];
    _Float16* bDst = &Bs[buf][rowL][colL];
    #pragma unroll
    for (int i = 0; i < 4; ++i) {
      float4 f0 = aReg[2 * i], f1 = aReg[2 * i + 1];
      union { fp16x2 h[4]; uint4 q; } ua;
      ua.h[0] = __builtin_amdgcn_cvt_pkrtz(f0.x, f0.y);   // v_cvt_pkrtz_f16_f32
      ua.h[1] = __builtin_amdgcn_cvt_pkrtz(f0.z, f0.w);
      ua.h[2] = __builtin_amdgcn_cvt_pkrtz(f1.x, f1.y);
      ua.h[3] = __builtin_amdgcn_cvt_pkrtz(f1.z, f1.w);
      *(uint4*)(aDst + i * 8) = ua.q;                    // ds_store_b128
      int4 c0 = bReg[2 * i], c1 = bReg[2 * i + 1];
      union { fp16x2 h[4]; uint4 q; } ub;
      // codes 0..255 are exact in f16; scale+offset via packed f16 FMA (v_pk_fma_f16)
      ub.h[0] = __builtin_amdgcn_cvt_pkrtz((float)c0.x, (float)c0.y) * sc2 + ofs2;
      ub.h[1] = __builtin_amdgcn_cvt_pkrtz((float)c0.z, (float)c0.w) * sc2 + ofs2;
      ub.h[2] = __builtin_amdgcn_cvt_pkrtz((float)c1.x, (float)c1.y) * sc2 + ofs2;
      ub.h[3] = __builtin_amdgcn_cvt_pkrtz((float)c1.z, (float)c1.w) * sc2 + ofs2;
      *(uint4*)(bDst + i * 8) = ub.q;                    // ds_store_b128
    }
  };

  auto compute = [&](int buf) {
    #pragma unroll
    for (int kk = 0; kk < BK; kk += 32) {
      // B fragments (K x N): row-major weight tile in LDS == ISA B layout
      half16v bF[2];
      #pragma unroll
      for (int j = 0; j < 2; ++j) {
        const _Float16* p = &Bs[buf][wnBase + j * 16 + laneLo][kk + laneHi8];
        union { uint4 q[2]; half16v v; } u;
        u.q[0] = *(const uint4*)p;                // K 0..7  (rel, per lane group)
        u.q[1] = *(const uint4*)(p + 16);         // K 16..23 (rel)
        bF[j] = u.v;
      }
      #pragma unroll
      for (int i = 0; i < 4; ++i) {
        const _Float16* p = &As[buf][wmBase + i * 16 + laneLo][kk + laneHi8];
        union { uint4 q[2]; half16v v; } u;
        u.q[0] = *(const uint4*)p;
        u.q[1] = *(const uint4*)(p + 16);
        half16v aF = u.v;
        #pragma unroll
        for (int j = 0; j < 2; ++j)
          acc[i][j] = __builtin_amdgcn_wmma_f32_16x16x32_f16(
              false, aF, false, bF[j], (short)0, acc[i][j], false, false);
      }
    }
  };

  // software pipeline: reg-prefetch next stage while WMMAs run on current LDS buf
  const int nStages = K / BK;
  g2r(0);
  r2lds(0);
  __syncthreads();
  for (int s = 0; s < nStages; ++s) {
    const int buf = s & 1;
    if (s + 1 < nStages) g2r(s + 1);
    compute(buf);
    __syncthreads();
    if (s + 1 < nStages) {
      r2lds(buf ^ 1);
      __syncthreads();
    }
  }

  // epilogue: 16x16 f32 C/D layout -> coalesced non-temporal b32 stores
  #pragma unroll
  for (int i = 0; i < 4; ++i) {
    #pragma unroll
    for (int j = 0; j < 2; ++j) {
      const int mBase = m0 + wmBase + i * 16 + (laneHi8 >> 3) * 8;  // +0 or +8
      const int nCol  = n0 + wnBase + j * 16 + laneLo;
      float* o = out + (size_t)mBase * N + nCol;
      #pragma unroll
      for (int r = 0; r < 8; ++r)
        __builtin_nontemporal_store(acc[i][j][r], o + (size_t)r * N);
    }
  }
}

extern "C" void kernel_launch(void* const* d_in, const int* in_sizes, int n_in,
                              void* d_out, int out_size, void* d_ws, size_t ws_size,
                              hipStream_t stream) {
  const float* x   = (const float*)d_in[0];
  const int*   qw  = (const int*)d_in[1];
  const float* am  = (const float*)d_in[2];
  float*       out = (float*)d_out;

  // derive dims: |x| = M*K, |qw| = V*K, out_size = M*V  =>  K^2 = |x|*|qw|/out
  const long long xsz = in_sizes[0];
  const long long qsz = in_sizes[1];
  const long long d2  = (xsz * qsz) / (long long)out_size;
  const int K = (int)llround(sqrt((double)d2));   // 2048
  const int M = (int)(xsz / K);                   // 4096
  const int V = (int)(qsz / K);                   // 32000

  dim3 grid(M / BM, V / BN);   // (32, 250); M fast -> weight-stripe L2 reuse
  dim3 block(256);
  hipLaunchKernelGGL(qgemm_wmma_f16, grid, block, 0, stream,
                     x, qw, am, out, M, V, K);
}
